// Attention_84043920048285
// MI455X (gfx1250) — matrix operand order repaired and checked by reading, not compiled, not run on hardware
//
#include <hip/hip_runtime.h>

typedef __attribute__((ext_vector_type(16))) _Float16 v16h;
typedef __attribute__((ext_vector_type(8)))  float    v8f;
typedef __attribute__((ext_vector_type(4)))  float    f4;

#define D_MODEL 2048
#define NUM_Q   8
#define KV      4096
#define NKEYS   (KV + 1)      // 4097 keys after cache append
#define SROW    4112          // padded score-row stride (16-float aligned)
#define SCALE   0.08838834764831845f   // 1/sqrt(128)

__device__ __forceinline__ void cvt4(v16h& dst, int base, f4 v) {
    dst[base + 0] = (_Float16)v.x;
    dst[base + 1] = (_Float16)v.y;
    dst[base + 2] = (_Float16)v.z;
    dst[base + 3] = (_Float16)v.w;
}

// ---------------------------------------------------------------------------
// Pass 1: scores[b][q][key] = SCALE * sum_d Q[b][q][d] * K[b][key][d]
// grid (ceil(4097/128), B), 256 threads (8 waves); each wave: 16 keys.
// WMMA tile: M=16 (8 real queries + 8 dup rows, discarded), N=16 keys,
// K-step=32 over d. A loads are unconditional: padding rows replicate query 0
// and their D rows are never stored, so no divergence in the hot loop.
// ---------------------------------------------------------------------------
__global__ __launch_bounds__(256)
void qk_kernel(const float* __restrict__ q,
               const float* __restrict__ k_new,
               const float* __restrict__ k_cache,
               float* __restrict__ scores)
{
    const int lane = threadIdx.x & 31;
    const int wave = threadIdx.x >> 5;
    const int b    = blockIdx.y;
    const int n    = lane & 15;          // B-matrix column (key) AND A-matrix row (query)
    const int hl   = lane >> 4;          // half-lane select
    const int key  = (blockIdx.x * 8 + wave) * 16 + n;

    // Row of K for this lane's key (key==4096 -> k_new; key>4096 -> dummy, never stored)
    const float* krow = (key < KV) ? (k_cache + ((size_t)b * KV + key) * D_MODEL)
                                   : (k_new + (size_t)b * D_MODEL);
    const int m = n;                      // query row; rows >=8 clamp to row 0 (discarded)
    const float* qrow = q + ((size_t)b * NUM_Q + (m < NUM_Q ? m : 0)) * D_MODEL;

    v8f c = {};
    for (int d0 = 0; d0 < D_MODEL; d0 += 32) {
        // A fragment (16x32 f16): elem e -> d0+16*hl+e
        const float* qa = qrow + d0 + 16 * hl;
        v16h a;
        cvt4(a, 0,  *(const f4*)(qa + 0));
        cvt4(a, 4,  *(const f4*)(qa + 4));
        cvt4(a, 8,  *(const f4*)(qa + 8));
        cvt4(a, 12, *(const f4*)(qa + 12));

        // B fragment (32x16 f16): elem e -> d = d0 + 16*hl + e (contiguous 64B/lane).
        // K stream is touched exactly once device-wide -> non-temporal.
        const float* kb = krow + d0 + 16 * hl;
        v16h bf;
        cvt4(bf, 0,  __builtin_nontemporal_load((const f4*)(kb + 0)));
        cvt4(bf, 4,  __builtin_nontemporal_load((const f4*)(kb + 4)));
        cvt4(bf, 8,  __builtin_nontemporal_load((const f4*)(kb + 8)));
        cvt4(bf, 12, __builtin_nontemporal_load((const f4*)(kb + 12)));

        c = __builtin_amdgcn_wmma_f32_16x16x32_f16(false, a, false, bf,
                                                   (short)0, c, false, false);
    }

    // D layout: elem e holds (m = e + 8*hl, n = lane&15). Real queries live on hl==0.
    if (hl == 0 && key <= KV) {
        float* srow = scores + (size_t)(b * NUM_Q) * SROW + key;
        #pragma unroll
        for (int e = 0; e < NUM_Q; e++)
            srow[(size_t)e * SROW] = c[e] * SCALE;
    }
}

// ---------------------------------------------------------------------------
// Pass 2: softmax over each score row (length 4097), in place.
// grid (B*8), 256 threads.
// ---------------------------------------------------------------------------
__global__ __launch_bounds__(256)
void softmax_kernel(float* __restrict__ scores)
{
    __shared__ float red[256];
    float* s = scores + (size_t)blockIdx.x * SROW;
    const int t = threadIdx.x;

    float m = -1e30f;
    for (int i = t; i < NKEYS; i += 256) m = fmaxf(m, s[i]);
    red[t] = m; __syncthreads();
    for (int off = 128; off > 0; off >>= 1) {
        if (t < off) red[t] = fmaxf(red[t], red[t + off]);
        __syncthreads();
    }
    m = red[0]; __syncthreads();

    float sum = 0.f;
    for (int i = t; i < NKEYS; i += 256) {
        float e = __expf(s[i] - m);
        s[i] = e;
        sum += e;
    }
    red[t] = sum; __syncthreads();
    for (int off = 128; off > 0; off >>= 1) {
        if (t < off) red[t] += red[t + off];
        __syncthreads();
    }
    const float inv = 1.0f / red[0];
    for (int i = t; i < NKEYS; i += 256) s[i] *= inv;
}

// ---------------------------------------------------------------------------
// Pass 3: out[b][q][d] = sum_k W[b][q][k] * V[b][k][d]
// grid (2048/128, B), 256 threads; each wave owns 16 features d.
// Main loop covers keys [0,4096) via WMMA; key 4096 (v_new) is a scalar
// rank-1 epilogue. V loads stay regular-temporal: all 16 d-tile WGs of a
// batch co-stream the same 32 MB of V, which L2 (192 MB) captures.
// ---------------------------------------------------------------------------
__global__ __launch_bounds__(256)
void wv_kernel(const float* __restrict__ scores,
               const float* __restrict__ v_new,
               const float* __restrict__ v_cache,
               float* __restrict__ out)
{
    const int lane = threadIdx.x & 31;
    const int wave = threadIdx.x >> 5;
    const int b    = blockIdx.y;
    const int n    = lane & 15;
    const int hl   = lane >> 4;
    const int m    = n;                                  // query row; >=8 clamp (discarded)
    const int d    = (blockIdx.x * 8 + wave) * 16 + n;   // output feature

    const float* wrow = scores + ((size_t)b * NUM_Q + (m < NUM_Q ? m : 0)) * SROW;
    const float* vb   = v_cache + (size_t)b * KV * D_MODEL + d;

    v8f c = {};
    for (int k0 = 0; k0 < KV; k0 += 32) {
        // A fragment: weights W[m][k0 .. k0+31] (max index 4095 < 4097, no guard)
        const float* wa = wrow + k0 + 16 * hl;
        v16h a;
        cvt4(a, 0,  *(const f4*)(wa + 0));
        cvt4(a, 4,  *(const f4*)(wa + 4));
        cvt4(a, 8,  *(const f4*)(wa + 8));
        cvt4(a, 12, *(const f4*)(wa + 12));

        // B fragment: elem e -> V[k0 + 16*hl + e][d]  (stride D_MODEL column slice)
        const float* vcol = vb + (size_t)(k0 + 16 * hl) * D_MODEL;
        v16h bf;
        #pragma unroll
        for (int e = 0; e < 16; e++)
            bf[e] = (_Float16)vcol[(size_t)e * D_MODEL];

        c = __builtin_amdgcn_wmma_f32_16x16x32_f16(false, a, false, bf,
                                                   (short)0, c, false, false);
    }

    // Epilogue: key 4096 contribution + store (real queries on hl==0, m=e)
    if (hl == 0) {
        const float vn = v_new[(size_t)b * D_MODEL + d];
        float* orow = out + (size_t)b * NUM_Q * D_MODEL + d;
        #pragma unroll
        for (int e = 0; e < NUM_Q; e++) {
            const float w = scores[((size_t)b * NUM_Q + e) * SROW + KV];
            orow[(size_t)e * D_MODEL] = c[e] + w * vn;
        }
    }
}

// ---------------------------------------------------------------------------
extern "C" void kernel_launch(void* const* d_in, const int* in_sizes, int n_in,
                              void* d_out, int out_size, void* d_ws, size_t ws_size,
                              hipStream_t stream)
{
    (void)in_sizes; (void)n_in; (void)out_size; (void)ws_size;
    const float* q       = (const float*)d_in[0];
    const float* k_new   = (const float*)d_in[1];
    const float* v_new   = (const float*)d_in[2];
    const float* k_cache = (const float*)d_in[3];
    const float* v_cache = (const float*)d_in[4];
    float* out    = (float*)d_out;
    float* scores = (float*)d_ws;   // needs 16*8*4112*4 B ~= 2.1 MB

    const int B = 16;
    dim3 blk(256);

    dim3 g1((NKEYS + 127) / 128, B);          // 33 x 16 workgroups
    qk_kernel<<<g1, blk, 0, stream>>>(q, k_new, k_cache, scores);

    softmax_kernel<<<dim3(B * NUM_Q), blk, 0, stream>>>(scores);

    dim3 g3(D_MODEL / 128, B);                // 16 x 16 workgroups
    wv_kernel<<<g3, blk, 0, stream>>>(scores, v_new, v_cache, out);
}
